// GlobalGuidedAoERouter_46574625358023
// MI455X (gfx1250) — matrix operand order repaired and checked by reading, not compiled, not run on hardware
//
#include <hip/hip_runtime.h>
#include <hip/hip_bf16.h>
#include <stdint.h>

// GlobalGuidedAoERouter for MI455X (gfx1250, wave32, WMMA).
// GEMMs: v_wmma_f32_16x16x32_f16, tiles staged via global_load_async_to_lds_b128
// (ASYNCcnt DMA, double-buffered). Small per-token work on VALU.

typedef __attribute__((ext_vector_type(16))) _Float16 v16h;
typedef __attribute__((ext_vector_type(8)))  _Float16 v8h;
typedef __attribute__((ext_vector_type(8)))  float    v8f;

#define NB    4
#define TT    2048
#define DD    1024
#define DLD   128
#define EE    8
#define HH    4
#define HDD   32
#define NN    (NB*TT)          // 8192
#define LSTR  40               // LDS row stride (f16) for 32-wide K slice + pad

__device__ __forceinline__ float gelu_exact(float x) {
  return 0.5f * x * (1.0f + erff(x * 0.70710678118654752f));
}

// ---------------------------------------------------------------------------
__global__ void k_convert_h(const float* __restrict__ s, _Float16* __restrict__ d, int n) {
  for (int i = blockIdx.x * blockDim.x + threadIdx.x; i < n; i += gridDim.x * blockDim.x)
    d[i] = (_Float16)s[i];
}

// fp32 (rows,cols) row-major -> f16 transposed (cols,rows)
__global__ void k_transpose_h(const float* __restrict__ s, _Float16* __restrict__ d,
                              int rows, int cols) {
  int idx = blockIdx.x * 256 + threadIdx.x;
  if (idx < rows * cols) {
    int r = idx / cols, c = idx % cols;
    d[(size_t)c * rows + r] = (_Float16)s[idx];
  }
}

// ---------------------------------------------------------------------------
// gctx = mean_T(x[b]) @ Wg^T + bg ; rq = gelu(LN(gctx@rq1^T+b)) @ rq2^T + b
__global__ __launch_bounds__(256) void k_gctx_rq(
    const float* __restrict__ x,
    const float* __restrict__ gw, const float* __restrict__ gb,
    const float* __restrict__ rq1w, const float* __restrict__ rq1b,
    const float* __restrict__ lnw, const float* __restrict__ lnb,
    const float* __restrict__ rq2w, const float* __restrict__ rq2b,
    float* __restrict__ gctx, float* __restrict__ rqv)
{
  __shared__ float xm[DD];
  __shared__ float gc[DLD];
  __shared__ float h1[DLD];
  __shared__ float mv[2];
  const int b = blockIdx.x, tid = threadIdx.x;
  const float* xb = x + (size_t)b * TT * DD;
  int c0 = tid * 4;
  float a0 = 0.f, a1 = 0.f, a2 = 0.f, a3 = 0.f;
  for (int t = 0; t < TT; ++t) {
    const float* r = xb + (size_t)t * DD + c0;
    a0 += r[0]; a1 += r[1]; a2 += r[2]; a3 += r[3];
  }
  const float invT = 1.0f / (float)TT;
  xm[c0] = a0 * invT; xm[c0+1] = a1 * invT; xm[c0+2] = a2 * invT; xm[c0+3] = a3 * invT;
  __syncthreads();
  if (tid < DLD) {
    float g = gb[tid];
    const float* wr = gw + (size_t)tid * DD;
    for (int k = 0; k < DD; ++k) g += xm[k] * wr[k];
    gc[tid] = g;
    gctx[b * DLD + tid] = g;
  }
  __syncthreads();
  if (tid < DLD) {
    float v = rq1b[tid];
    const float* wr = rq1w + tid * DLD;
    for (int k = 0; k < DLD; ++k) v += gc[k] * wr[k];
    h1[tid] = v;
  }
  __syncthreads();
  if (tid == 0) {
    float m = 0.f;
    for (int k = 0; k < DLD; ++k) m += h1[k];
    m /= DLD;
    float var = 0.f;
    for (int k = 0; k < DLD; ++k) { float d = h1[k] - m; var += d * d; }
    mv[0] = m; mv[1] = var / DLD;
  }
  __syncthreads();
  if (tid < DLD) {
    float hn = (h1[tid] - mv[0]) * rsqrtf(mv[1] + 1e-5f) * lnw[tid] + lnb[tid];
    gc[tid] = gelu_exact(hn);
  }
  __syncthreads();
  if (tid < DLD) {
    float v = rq2b[tid];
    const float* wr = rq2w + tid * DLD;
    for (int k = 0; k < DLD; ++k) v += gc[k] * wr[k];
    rqv[b * DLD + tid] = v;
  }
}

// ---------------------------------------------------------------------------
// WMMA GEMM: C(MxN) = A(MxK,f16,rm) @ Bt^T  where Bt is B-transposed (NxK,f16,rm).
// Tiles staged with global_load_async_to_lds_b128, double-buffered.
// Block = 256 threads = 8 waves; tile 128x128; K-step 32.
// M,N multiples of 128; K multiple of 32.
__global__ __launch_bounds__(256) void wmma_gemm_f16(
    const _Float16* __restrict__ A, const _Float16* __restrict__ Bt,
    const float* __restrict__ bias,
    float* __restrict__ Cf, _Float16* __restrict__ Ch,
    int M, int Nc, int Kc)
{
  __shared__ _Float16 lA[2][128 * LSTR];   // [row][k] for current 32-wide K slice
  __shared__ _Float16 lB[2][128 * LSTR];   // [col][k] (col-major tile, natural from Bt)
  const int rbase = blockIdx.y * 128;
  const int cbase = blockIdx.x * 128;
  const int tid  = threadIdx.x;
  const int wv   = tid >> 5;
  const int lane = tid & 31;
  const int l16  = lane & 15;
  const int hf   = lane >> 4;

  // async chunk map: 512 x 16B chunks/tile; this thread: rows r0 and r0+64, k-offset kk
  const int r0 = tid >> 2;            // 0..63
  const int kk = (tid & 3) * 8;       // f16 index inside 32-wide slice

  const uint64_t gA0 = (uint64_t)(uintptr_t)(A  + (size_t)(rbase + r0     ) * Kc + kk);
  const uint64_t gA1 = (uint64_t)(uintptr_t)(A  + (size_t)(rbase + r0 + 64) * Kc + kk);
  const uint64_t gB0 = (uint64_t)(uintptr_t)(Bt + (size_t)(cbase + r0     ) * Kc + kk);
  const uint64_t gB1 = (uint64_t)(uintptr_t)(Bt + (size_t)(cbase + r0 + 64) * Kc + kk);
  const uint32_t lAb = (uint32_t)(uintptr_t)(&lA[0][0]) + (uint32_t)((r0 * LSTR + kk) * 2);
  const uint32_t lBb = (uint32_t)(uintptr_t)(&lB[0][0]) + (uint32_t)((r0 * LSTR + kk) * 2);
  const uint32_t half64 = 64 * LSTR * 2;
  const uint32_t bufStride = 128 * LSTR * 2;

  v8f acc[8];
#pragma unroll
  for (int i = 0; i < 8; ++i)
#pragma unroll
    for (int j = 0; j < 8; ++j) acc[i][j] = 0.0f;

  auto issue_tile = [&](int kb, int buf) {
    const uint64_t go = (uint64_t)kb * 2;
    const uint32_t lo = buf ? bufStride : 0u;
    asm volatile("global_load_async_to_lds_b128 %0, %1, off"
                 :: "v"(lAb + lo), "v"(gA0 + go) : "memory");
    asm volatile("global_load_async_to_lds_b128 %0, %1, off"
                 :: "v"(lAb + lo + half64), "v"(gA1 + go) : "memory");
    asm volatile("global_load_async_to_lds_b128 %0, %1, off"
                 :: "v"(lBb + lo), "v"(gB0 + go) : "memory");
    asm volatile("global_load_async_to_lds_b128 %0, %1, off"
                 :: "v"(lBb + lo + half64), "v"(gB1 + go) : "memory");
  };

  const int ntiles = Kc >> 5;
  issue_tile(0, 0);
  for (int it = 0; it < ntiles; ++it) {
    asm volatile("s_wait_asynccnt 0x0" ::: "memory");
    __syncthreads();
    if (it + 1 < ntiles) issue_tile((it + 1) << 5, (it + 1) & 1);

    const _Float16* la = lA[it & 1];
    const _Float16* lb = lB[it & 1];

    // A fragment (16x32): lane<16 -> K 0..7 & 16..23; lane>=16 -> 8..15 & 24..31
    v16h af;
    {
      v8h a0 = *(const v8h*)&la[(wv * 16 + l16) * LSTR + hf * 8];
      v8h a1 = *(const v8h*)&la[(wv * 16 + l16) * LSTR + 16 + hf * 8];
#pragma unroll
      for (int j = 0; j < 8; ++j) { af[j] = a0[j]; af[8 + j] = a1[j]; }
    }
    // preload all 8 B fragments, then issue 8 WMMAs back-to-back
    v16h bfr[8];
#pragma unroll
    for (int nt = 0; nt < 8; ++nt) {
      v8h b0 = *(const v8h*)&lb[(nt * 16 + l16) * LSTR + hf * 8];
      v8h b1 = *(const v8h*)&lb[(nt * 16 + l16) * LSTR + 16 + hf * 8];
#pragma unroll
      for (int j = 0; j < 8; ++j) { bfr[nt][j] = b0[j]; bfr[nt][8 + j] = b1[j]; }
    }
#pragma unroll
    for (int nt = 0; nt < 8; ++nt)
      acc[nt] = __builtin_amdgcn_wmma_f32_16x16x32_f16(
          false, af, false, bfr[nt], (short)0, acc[nt], false, false);
  }

  // epilogue: C layout: VGPR i -> row i (lane<16) / i+8 (lane>=16), col = lane%16
#pragma unroll
  for (int nt = 0; nt < 8; ++nt) {
#pragma unroll
    for (int i = 0; i < 8; ++i) {
      int mrow = wv * 16 + ((lane < 16) ? i : i + 8);
      int mcol = nt * 16 + l16;
      size_t gi = (size_t)(rbase + mrow) * Nc + cbase + mcol;
      float v = acc[nt][i];
      if (bias) v += bias[cbase + mcol];
      if (Cf) Cf[gi] = v;
      if (Ch) Ch[gi] = (_Float16)v;
    }
  }
}

// ---------------------------------------------------------------------------
// seqh[n, 0] = gctx[b]; seqh[n, 1+e] = ef[n,e,:] + pos[e,:]
__global__ void k_build_seq(const float* __restrict__ gctx, const float* __restrict__ ef,
                            const float* __restrict__ pos, _Float16* __restrict__ seqh)
{
  int n = blockIdx.x, d = threadIdx.x;
  int b = n >> 11;   // / TT
  seqh[((size_t)n * 9) * DLD + d] = (_Float16)gctx[b * DLD + d];
#pragma unroll
  for (int e = 0; e < EE; ++e)
    seqh[((size_t)n * 9 + 1 + e) * DLD + d] =
        (_Float16)(ef[(size_t)n * DD + e * DLD + d] + pos[e * DLD + d]);
}

// ---------------------------------------------------------------------------
// 9-token attention per n: 8 n per block (one wave each).
__global__ __launch_bounds__(256) void k_attn(const _Float16* __restrict__ qkv,
                                              _Float16* __restrict__ oh)
{
  __shared__ _Float16 sQ[8 * 9 * 384];   // 55296 B
  const int nb = blockIdx.x;
  const uint32_t* gsrc = (const uint32_t*)(qkv + (size_t)nb * 8 * 9 * 384);
  uint32_t* ldst = (uint32_t*)sQ;
  for (int i = threadIdx.x; i < 8 * 9 * 384 / 2; i += 256) ldst[i] = gsrc[i];
  __syncthreads();

  const int w = threadIdx.x >> 5, lane = threadIdx.x & 31;
  const _Float16* base = sQ + w * (9 * 384);
  const int n = nb * 8 + w;
  for (int t = lane; t < HH * 9; t += 32) {       // 36 (head,q) tasks
    int h = t / 9, q = t % 9;
    const _Float16* qp = base + q * 384 + h * HDD;
    float s[9];
    float mx = -1e30f;
#pragma unroll
    for (int kk = 0; kk < 9; ++kk) {
      const _Float16* kp = base + kk * 384 + DLD + h * HDD;
      float d = 0.f;
#pragma unroll
      for (int j = 0; j < HDD; ++j) d += (float)qp[j] * (float)kp[j];
      s[kk] = d * 0.17677669529663687f;   // 1/sqrt(32)
      mx = fmaxf(mx, s[kk]);
    }
    float den = 0.f;
#pragma unroll
    for (int kk = 0; kk < 9; ++kk) { s[kk] = expf(s[kk] - mx); den += s[kk]; }
    float inv = 1.0f / den;
    float o[HDD];
#pragma unroll
    for (int j = 0; j < HDD; ++j) o[j] = 0.f;
#pragma unroll
    for (int kk = 0; kk < 9; ++kk) {
      float p = s[kk] * inv;
      const _Float16* vp = base + kk * 384 + 2 * DLD + h * HDD;
#pragma unroll
      for (int j = 0; j < HDD; ++j) o[j] += p * (float)vp[j];
    }
    _Float16* op = oh + ((size_t)n * 9 + q) * DLD + h * HDD;
#pragma unroll
    for (int j = 0; j < HDD; ++j) op[j] = (_Float16)o[j];
  }
}

// ---------------------------------------------------------------------------
// refinedh[row=n*8+e] = LN(of[n, e+1] + (ef[n,e]+pos[e])) * nw + nb
__global__ __launch_bounds__(128) void k_ln(const float* __restrict__ of,
                                            const float* __restrict__ ef,
                                            const float* __restrict__ pos,
                                            const float* __restrict__ nw,
                                            const float* __restrict__ nbv,
                                            _Float16* __restrict__ refinedh)
{
  __shared__ float red[DLD];
  int row = blockIdx.x;
  int n = row >> 3, e = row & 7;
  int d = threadIdx.x;
  float xv = of[((size_t)n * 9 + e + 1) * DLD + d]
           + ef[(size_t)n * DD + e * DLD + d] + pos[e * DLD + d];
  red[d] = xv;
  __syncthreads();
  for (int s = 64; s > 0; s >>= 1) { if (d < s) red[d] += red[d + s]; __syncthreads(); }
  float m = red[0] / DLD;
  __syncthreads();
  float dv = xv - m;
  red[d] = dv * dv;
  __syncthreads();
  for (int s = 64; s > 0; s >>= 1) { if (d < s) red[d] += red[d + s]; __syncthreads(); }
  float var = red[0] / DLD;
  float y = (xv - m) * rsqrtf(var + 1e-5f) * nw[d] + nbv[d];
  refinedh[(size_t)row * DLD + d] = (_Float16)y;
}

// ---------------------------------------------------------------------------
// router per n: logits, softmax, top-3, Ahat = w_sel * gelu(ef); aux atomics
__global__ __launch_bounds__(256) void k_router(const float* __restrict__ rqv,
                                                const float* __restrict__ rkf,
                                                const float* __restrict__ ef,
                                                _Float16* __restrict__ Ahat,
                                                float* __restrict__ accP,
                                                float* __restrict__ accL)
{
  __shared__ float sl[EE];
  __shared__ float wsel[EE];
  const int n = blockIdx.x, b = n >> 11;
  const int w = threadIdx.x >> 5, lane = threadIdx.x & 31;
  float p = 0.f;
  for (int j = lane; j < DLD; j += 32)
    p += rqv[b * DLD + j] * rkf[((size_t)n * EE + w) * DLD + j];
  for (int off = 16; off > 0; off >>= 1) p += __shfl_down(p, off, 32);
  if (lane == 0) sl[w] = p * 0.08838834764831845f;   // 1/sqrt(128)
  __syncthreads();
  if (threadIdx.x == 0) {
    float mx = -1e30f;
    for (int e = 0; e < EE; ++e) mx = fmaxf(mx, sl[e]);
    float ex[EE], ssum = 0.f;
    for (int e = 0; e < EE; ++e) { ex[e] = expf(sl[e] - mx); ssum += ex[e]; }
    float pr[EE];
    for (int e = 0; e < EE; ++e) pr[e] = ex[e] / ssum;
    bool used[EE];
    for (int e = 0; e < EE; ++e) used[e] = false;
    float sw = 0.f;
    for (int kk = 0; kk < 3; ++kk) {
      int bi = 0; float bv = -1e30f;
      for (int e = 0; e < EE; ++e)
        if (!used[e] && pr[e] > bv) { bv = pr[e]; bi = e; }
      used[bi] = true; sw += bv;
    }
    for (int e = 0; e < EE; ++e) {
      wsel[e] = used[e] ? pr[e] / sw : 0.f;
      atomicAdd(&accP[e], pr[e]);
      if (used[e]) atomicAdd(&accL[e], 1.0f);
    }
  }
  __syncthreads();
  for (int i = threadIdx.x; i < DD; i += 256) {
    int e = i >> 7;
    float g = ef[(size_t)n * DD + i];
    Ahat[(size_t)n * DD + i] = (_Float16)(wsel[e] * gelu_exact(g));
  }
}

__global__ void k_aux(const float* __restrict__ accP, const float* __restrict__ accL,
                      float* __restrict__ out)
{
  if (threadIdx.x == 0) {
    float s = 0.f;
    const float invN = 1.0f / (float)NN;
    for (int e = 0; e < EE; ++e) s += (accP[e] * invN) * (accL[e] * invN);
    out[0] = (float)EE * s;
  }
}

// ---------------------------------------------------------------------------
extern "C" void kernel_launch(void* const* d_in, const int* in_sizes, int n_in,
                              void* d_out, int out_size, void* d_ws, size_t ws_size,
                              hipStream_t stream)
{
  (void)in_sizes; (void)n_in; (void)out_size; (void)ws_size;
  const float* x    = (const float*)d_in[0];
  const float* wdn  = (const float*)d_in[1];
  const float* pos  = (const float*)d_in[2];
  const float* gpw  = (const float*)d_in[3];
  const float* gpb  = (const float*)d_in[4];
  const float* aiw  = (const float*)d_in[5];
  const float* aib  = (const float*)d_in[6];
  const float* aow  = (const float*)d_in[7];
  const float* aob  = (const float*)d_in[8];
  const float* nw   = (const float*)d_in[9];
  const float* nbv  = (const float*)d_in[10];
  const float* rq1w = (const float*)d_in[11];
  const float* rq1b = (const float*)d_in[12];
  const float* lnw  = (const float*)d_in[13];
  const float* lnb  = (const float*)d_in[14];
  const float* rq2w = (const float*)d_in[15];
  const float* rq2b = (const float*)d_in[16];
  const float* rkw  = (const float*)d_in[17];
  const float* rkb  = (const float*)d_in[18];
  const float* wup  = (const float*)d_in[19];

  char* ws = (char*)d_ws;
  size_t cur = 0;
  auto alloc = [&](size_t bytes) -> char* {
    char* p = ws + cur;
    cur = (cur + bytes + 255) & ~(size_t)255;
    return p;
  };
  // weights as f16, stored as B^T (N x K) row-major — for w_down/attn_in/attn_out/rk
  // that is the ORIGINAL layout, so plain converts; only w_up needs a transpose.
  _Float16* wdh  = (_Float16*)alloc((size_t)1024 * 1024 * 2);   // (out=1024, k=1024)
  _Float16* aih  = (_Float16*)alloc((size_t)384 * 128 * 2);     // (out=384, k=128)
  _Float16* aoh  = (_Float16*)alloc((size_t)128 * 128 * 2);
  _Float16* rkh  = (_Float16*)alloc((size_t)128 * 128 * 2);
  _Float16* wupT = (_Float16*)alloc((size_t)1024 * 1024 * 2);   // (m=1024, k=1024)
  float* gctx = (float*)alloc(NB * DLD * 4);
  float* rqv  = (float*)alloc(NB * DLD * 4);
  float* accP = (float*)alloc(16 * 4);   // accP[0..7], accL[0..7]
  float* accL = accP + 8;
  char* regA  = alloc((size_t)NN * DD * 2);        // xh -> refinedh -> Ahat
  float* ef   = (float*)alloc((size_t)NN * DD * 4);
  char* regC  = alloc((size_t)NN * 9 * DLD * 2);   // seqh -> oh
  char* regD  = alloc((size_t)NN * 9 * 384 * 2);   // qkvh -> of(f32) -> rkf(f32)

  _Float16* xh       = (_Float16*)regA;
  _Float16* refinedh = (_Float16*)regA;
  _Float16* Ahat     = (_Float16*)regA;
  _Float16* seqh     = (_Float16*)regC;
  _Float16* oh       = (_Float16*)regC;
  _Float16* qkvh     = (_Float16*)regD;
  float*    of       = (float*)regD;
  float*    rkf      = (float*)regD;

  hipMemsetAsync(accP, 0, 16 * 4, stream);

  // precision conversions (+ single transpose for w_up)
  k_convert_h  <<<2048, 256, 0, stream>>>(x,   xh,   NN * DD);
  k_convert_h  <<<1024, 256, 0, stream>>>(wdn, wdh,  1024 * 1024);
  k_convert_h  <<<192,  256, 0, stream>>>(aiw, aih,  384 * 128);
  k_convert_h  <<<64,   256, 0, stream>>>(aow, aoh,  128 * 128);
  k_convert_h  <<<64,   256, 0, stream>>>(rkw, rkh,  128 * 128);
  k_transpose_h<<<4096, 256, 0, stream>>>(wup, wupT, 1024, 1024);

  // global context + refined query (tiny)
  k_gctx_rq<<<NB, 256, 0, stream>>>(x, gpw, gpb, rq1w, rq1b, lnw, lnb,
                                    rq2w, rq2b, gctx, rqv);

  // expert_feats = x @ w_down^T          (8192 x 1024 x 1024)
  wmma_gemm_f16<<<dim3(8, 64), 256, 0, stream>>>(xh, wdh, nullptr, ef, nullptr,
                                                 NN, 1024, 1024);
  // seq tokens (f16)
  k_build_seq<<<NN, 128, 0, stream>>>(gctx, ef, pos, seqh);

  // qkv = seq @ attn_in^T + b            (73728 x 384 x 128)
  wmma_gemm_f16<<<dim3(3, 576), 256, 0, stream>>>(seqh, aih, aib, nullptr, qkvh,
                                                  NN * 9, 384, 128);
  // per-n 9-token attention
  k_attn<<<NN / 8, 256, 0, stream>>>(qkvh, oh);

  // o = attn_o @ attn_out^T + b          (73728 x 128 x 128)
  wmma_gemm_f16<<<dim3(1, 576), 256, 0, stream>>>(oh, aoh, aob, of, nullptr,
                                                  NN * 9, 128, 128);
  // LN(o + seq) -> refined (tokens 1..8 only)
  k_ln<<<NN * EE, 128, 0, stream>>>(of, ef, pos, nw, nbv, refinedh);

  // rk = refined @ rk^T + b              (65536 x 128 x 128)
  wmma_gemm_f16<<<dim3(1, 512), 256, 0, stream>>>(refinedh, rkh, rkb, rkf, nullptr,
                                                  NN * EE, 128, 128);
  // router: logits/softmax/top-3 + Ahat = w_sel * gelu(ef)
  k_router<<<NN, 256, 0, stream>>>(rqv, rkf, ef, Ahat, accP, accL);

  // weighted = Ahat @ w_up_flat          (8192 x 1024 x 1024) -> d_out
  wmma_gemm_f16<<<dim3(8, 64), 256, 0, stream>>>(Ahat, wupT, nullptr,
                                                 (float*)d_out, nullptr,
                                                 NN, 1024, 1024);
  // aux loss -> d_out[N*D]
  k_aux<<<1, 32, 0, stream>>>(accP, accL, (float*)d_out + (size_t)NN * DD);
}
